// GCNWithPositionalEncoding_5909874999433
// MI455X (gfx1250) — matrix operand order repaired and checked by reading, not compiled, not run on hardware
//
#include <hip/hip_runtime.h>
#include <hip/hip_bf16.h>

// ---------------------------------------------------------------------------
// Types
// ---------------------------------------------------------------------------
typedef __bf16 bf16;
typedef bf16  v16bf __attribute__((ext_vector_type(16)));
typedef bf16  bf8v  __attribute__((ext_vector_type(8)));
typedef bf16  bf4v  __attribute__((ext_vector_type(4)));
typedef float v8f   __attribute__((ext_vector_type(8)));

#define N_NODES   50000
#define N_EDGES   600000
#define N_GRAPHS  64
#define EMB       256
#define NF_DIM    320      // 256 + 32 + 32
#define H1_DIM    128
#define H2_DIM    128
#define MAX_DEPTH 50
#define MAX_CHILD 20

// ---------------------------------------------------------------------------
// Weight swizzle: f32 [K,N] row-major -> WMMA B-operand tiles, bf16.
// Tile (nt,kt) at ((nt*Ktiles+kt) << 9) elements; per lane 16 contiguous bf16:
//   element i of lane l -> B[kt*32 + (l<16?0:16) + i][nt*16 + (l&15)]
// ---------------------------------------------------------------------------
__global__ void swizzle_w_kernel(const float* __restrict__ W, bf16* __restrict__ out,
                                 int K, int N) {
    int Ktiles = K >> 5;
    int Ntiles = N >> 4;
    int total  = Ktiles * Ntiles * 32;
    int tid = blockIdx.x * blockDim.x + threadIdx.x;
    if (tid >= total) return;
    int lane = tid & 31;
    int tile = tid >> 5;
    int kt = tile % Ktiles;
    int nt = tile / Ktiles;
    int n  = nt * 16 + (lane & 15);
    int kb = kt * 32 + ((lane >> 4) << 4);
    bf16* dst = out + ((size_t)tile << 9) + lane * 16;
#pragma unroll
    for (int i = 0; i < 16; ++i)
        dst[i] = (bf16)W[(size_t)(kb + i) * N + n];
}

// ---------------------------------------------------------------------------
// Embedding gather + concat -> nf bf16 [N_NODES, 320] row-major
// ---------------------------------------------------------------------------
__global__ void gather_nf_kernel(const int* __restrict__ x,
                                 const int* __restrict__ depth,
                                 const int* __restrict__ child,
                                 const float* __restrict__ node_emb,
                                 const float* __restrict__ depth_emb,
                                 const float* __restrict__ child_emb,
                                 bf16* __restrict__ nf) {
    size_t idx = (size_t)blockIdx.x * blockDim.x + threadIdx.x;   // N * 80 groups of 4
    if (idx >= (size_t)N_NODES * (NF_DIM / 4)) return;
    int i = (int)(idx / (NF_DIM / 4));
    int f = (int)(idx % (NF_DIM / 4)) * 4;
    float4 v;
    if (f < EMB) {
        v = *(const float4*)(node_emb + (size_t)x[i] * EMB + f);
    } else if (f < EMB + 32) {
        int d = depth[i]; d = d < 0 ? 0 : (d > MAX_DEPTH ? MAX_DEPTH : d);
        v = *(const float4*)(depth_emb + (size_t)d * 32 + (f - EMB));
    } else {
        int c = child[i]; c = c < 0 ? 0 : (c > MAX_CHILD ? MAX_CHILD : c);
        v = *(const float4*)(child_emb + (size_t)c * 32 + (f - EMB - 32));
    }
    bf4v o = { (bf16)v.x, (bf16)v.y, (bf16)v.z, (bf16)v.w };
    *(bf4v*)(nf + (size_t)i * NF_DIM + f) = o;
}

// ---------------------------------------------------------------------------
// Degree / normalization
// ---------------------------------------------------------------------------
__global__ void deg_init_kernel(float* __restrict__ deg) {
    int i = blockIdx.x * blockDim.x + threadIdx.x;
    if (i < N_NODES) deg[i] = 1.0f;   // self loop
}
__global__ void deg_edges_kernel(const int* __restrict__ dst, float* __restrict__ deg) {
    int e = blockIdx.x * blockDim.x + threadIdx.x;
    if (e < N_EDGES) atomicAdd(&deg[dst[e]], 1.0f);
}
__global__ void deg_finish_kernel(float* __restrict__ dinv) {
    int i = blockIdx.x * blockDim.x + threadIdx.x;
    if (i < N_NODES) dinv[i] = rsqrtf(dinv[i]);
}
// Per-edge norm precompute: norm[e] = dinv[src]*dinv[dst]
__global__ void edge_norm_kernel(const int* __restrict__ src, const int* __restrict__ dst,
                                 const float* __restrict__ dinv, float* __restrict__ norm) {
    int e = blockIdx.x * blockDim.x + threadIdx.x;
    if (e < N_EDGES) norm[e] = dinv[src[e]] * dinv[dst[e]];
}

// ---------------------------------------------------------------------------
// Generic bf16 WMMA GEMM: C[M,Ncols] = A[M,K] * B (pre-swizzled).
// One wave -> 16 (M) x 64 (N) tile block (4 WMMA n-tiles, shared A regs).
// mode 0: store f32 raw.  mode 1: bias + ReLU -> bf16 store.
// Requires: M % 16 == 0 (wave-uniform exit), K % 32 == 0, Ncols % 64 == 0.
// ---------------------------------------------------------------------------
__global__ __launch_bounds__(256)
void gemm_bf16_kernel(const bf16* __restrict__ A,
                      const bf16* __restrict__ Bsw,
                      float* __restrict__ outF,
                      bf16* __restrict__ outH,
                      const float* __restrict__ bias,
                      int M, int K, int Ncols, int mode) {
    const int lane  = threadIdx.x & 31;
    const int wave  = threadIdx.x >> 5;
    const int mtile = blockIdx.x * 8 + wave;
    if (mtile * 16 >= M) return;                 // wave-uniform -> EXEC stays full

    const int Ktiles = K >> 5;
    const int ntBase = blockIdx.y * 4;

    const int arow   = mtile * 16 + (lane & 15);
    const int kAoff  = (lane < 16) ? 0 : 16;     // lanes 16-31 take k 16..31 (contiguous order)
    const bf16* Arow = A + (size_t)arow * K;

    v8f acc[4];
#pragma unroll
    for (int t = 0; t < 4; ++t) acc[t] = (v8f){0.f,0.f,0.f,0.f,0.f,0.f,0.f,0.f};

    for (int kt = 0; kt < Ktiles; ++kt) {
        const int kbase = kt << 5;
        union { v16bf v; bf8v h[2]; } a;
        a.h[0] = *(const bf8v*)(Arow + kbase + kAoff);        // K {0..7 | 16..23}
        a.h[1] = *(const bf8v*)(Arow + kbase + kAoff + 8);    // K {8..15 | 24..31}
#pragma unroll
        for (int t = 0; t < 4; ++t) {
            const bf16* bp = Bsw + (((size_t)(ntBase + t) * Ktiles + kt) << 9) + lane * 16;
            union { v16bf v; bf8v h[2]; } b;
            b.h[0] = *(const bf8v*)(bp);
            b.h[1] = *(const bf8v*)(bp + 8);
            acc[t] = __builtin_amdgcn_wmma_f32_16x16x32_bf16(
                         false, a.v, false, b.v, (short)0, acc[t], false, false);
        }
    }

    // Epilogue. C/D layout: lane l col = l&15 ; vgpr r row = r + (l<16?0:8).
    const int ncol0 = lane & 15;
    const int rbase = mtile * 16 + ((lane >> 4) << 3);
#pragma unroll
    for (int t = 0; t < 4; ++t) {
        const int col = (ntBase + t) * 16 + ncol0;
        const float bv = (mode == 1) ? bias[col] : 0.0f;
#pragma unroll
        for (int r = 0; r < 8; ++r) {
            float v = acc[t][r];
            const size_t o = (size_t)(rbase + r) * Ncols + col;
            if (mode == 1) {
                v += bv;
                v = v > 0.0f ? v : 0.0f;
                outH[o] = (bf16)v;
            } else {
                outF[o] = v;
            }
        }
    }
}

// ---------------------------------------------------------------------------
// Zero fill (float4 granularity; n4 = count of float4s)
// ---------------------------------------------------------------------------
__global__ void zero_f32v4_kernel(float4* __restrict__ p, size_t n4) {
    size_t i = (size_t)blockIdx.x * blockDim.x + threadIdx.x;
    if (i < n4) p[i] = make_float4(0.f, 0.f, 0.f, 0.f);
}
__global__ void zero_f32_kernel(float* __restrict__ p, size_t n) {
    size_t i = (size_t)blockIdx.x * blockDim.x + threadIdx.x;
    if (i < n) p[i] = 0.0f;
}

// ---------------------------------------------------------------------------
// Scatter aggregation: agg[dst] += t[src] * norm[e].
// One thread per (edge, 4-feature group): b128 load + 4 L2 atomics.
// ---------------------------------------------------------------------------
__global__ void edge_agg_kernel(const int* __restrict__ src,
                                const int* __restrict__ dst,
                                const float* __restrict__ norm,
                                const float4* __restrict__ t,   // [N_NODES, 32] float4
                                float* __restrict__ agg) {
    size_t idx = (size_t)blockIdx.x * blockDim.x + threadIdx.x;
    if (idx >= (size_t)N_EDGES * 32) return;
    int e = (int)(idx >> 5);
    int q = (int)(idx & 31);
    int s = src[e], d = dst[e];
    float nrm = norm[e];
    float4 v = t[(size_t)s * 32 + q];
    float* ap = agg + (size_t)d * 128 + q * 4;
    atomicAdd(ap + 0, v.x * nrm);
    atomicAdd(ap + 1, v.y * nrm);
    atomicAdd(ap + 2, v.z * nrm);
    atomicAdd(ap + 3, v.w * nrm);
}

// finalize layer 1: h1 = bf16(relu(agg + t*dinv^2 + b))   (self-loop folded in)
__global__ void finalize_relu_bf16_kernel(const float4* __restrict__ agg,
                                          const float4* __restrict__ t,
                                          const float* __restrict__ dinv,
                                          const float4* __restrict__ bias,
                                          bf16* __restrict__ outH) {
    size_t idx = (size_t)blockIdx.x * blockDim.x + threadIdx.x;   // N * 32
    if (idx >= (size_t)N_NODES * 32) return;
    int i = (int)(idx >> 5);
    int q = (int)(idx & 31);
    float di = dinv[i];
    float d2 = di * di;
    float4 a = agg[idx], tv = t[idx], b = bias[q];
    float vx = a.x + tv.x * d2 + b.x;
    float vy = a.y + tv.y * d2 + b.y;
    float vz = a.z + tv.z * d2 + b.z;
    float vw = a.w + tv.w * d2 + b.w;
    vx = vx > 0.f ? vx : 0.f;  vy = vy > 0.f ? vy : 0.f;
    vz = vz > 0.f ? vz : 0.f;  vw = vw > 0.f ? vw : 0.f;
    bf4v o = { (bf16)vx, (bf16)vy, (bf16)vz, (bf16)vw };
    *(bf4v*)(outH + (size_t)i * 128 + q * 4) = o;                 // b64 store
}

// finalize layer 2 + mean-pool numerator: pooled[batch[i]] += relu(...)
__global__ void finalize_pool_kernel(const float4* __restrict__ agg,
                                     const float4* __restrict__ t,
                                     const float* __restrict__ dinv,
                                     const float4* __restrict__ bias,
                                     const int* __restrict__ batch,
                                     float* __restrict__ pooled) {
    size_t idx = (size_t)blockIdx.x * blockDim.x + threadIdx.x;   // N * 32
    if (idx >= (size_t)N_NODES * 32) return;
    int i = (int)(idx >> 5);
    int q = (int)(idx & 31);
    float di = dinv[i];
    float d2 = di * di;
    float4 a = agg[idx], tv = t[idx], b = bias[q];
    float vx = a.x + tv.x * d2 + b.x;
    float vy = a.y + tv.y * d2 + b.y;
    float vz = a.z + tv.z * d2 + b.z;
    float vw = a.w + tv.w * d2 + b.w;
    vx = vx > 0.f ? vx : 0.f;  vy = vy > 0.f ? vy : 0.f;
    vz = vz > 0.f ? vz : 0.f;  vw = vw > 0.f ? vw : 0.f;
    float* pp = pooled + (size_t)batch[i] * 128 + q * 4;
    atomicAdd(pp + 0, vx);
    atomicAdd(pp + 1, vy);
    atomicAdd(pp + 2, vz);
    atomicAdd(pp + 3, vw);
}

__global__ void count_nodes_kernel(const int* __restrict__ batch, float* __restrict__ count) {
    int i = blockIdx.x * blockDim.x + threadIdx.x;
    if (i < N_NODES) atomicAdd(&count[batch[i]], 1.0f);
}

// out[g] = dot(pooled[g]/max(count,1), clf_W) + clf_b
__global__ void classify_kernel(const float* __restrict__ pooled,
                                const float* __restrict__ count,
                                const float* __restrict__ clfW,
                                const float* __restrict__ clfb,
                                float* __restrict__ out) {
    int g = threadIdx.x;
    if (g >= N_GRAPHS) return;
    float c = count[g];
    c = c > 1.0f ? c : 1.0f;
    float inv = 1.0f / c;
    float s = 0.0f;
    for (int f = 0; f < 128; ++f)
        s += pooled[(size_t)g * 128 + f] * inv * clfW[f];
    out[g] = s + clfb[0];
}

// ---------------------------------------------------------------------------
// Launch
// ---------------------------------------------------------------------------
extern "C" void kernel_launch(void* const* d_in, const int* in_sizes, int n_in,
                              void* d_out, int out_size, void* d_ws, size_t ws_size,
                              hipStream_t stream) {
    (void)in_sizes; (void)n_in; (void)out_size; (void)ws_size;

    const int*   x         = (const int*)  d_in[0];
    const int*   edges     = (const int*)  d_in[1];   // [2, E] row-major
    const int*   batch     = (const int*)  d_in[2];
    const int*   depth     = (const int*)  d_in[3];
    const int*   child     = (const int*)  d_in[4];
    const float* node_emb  = (const float*)d_in[5];
    const float* depth_emb = (const float*)d_in[6];
    const float* child_emb = (const float*)d_in[7];
    const float* proj_W    = (const float*)d_in[8];
    const float* proj_b    = (const float*)d_in[9];
    const float* W1        = (const float*)d_in[10];
    const float* b1        = (const float*)d_in[11];
    const float* W2        = (const float*)d_in[12];
    const float* b2        = (const float*)d_in[13];
    const float* clf_W     = (const float*)d_in[14];
    const float* clf_b     = (const float*)d_in[15];
    float* out = (float*)d_out;

    const int* e_src = edges;
    const int* e_dst = edges + N_EDGES;

    // -------- workspace carve-out (256B aligned regions) --------
    char*  ws  = (char*)d_ws;
    size_t off = 0;
    auto carve = [&](size_t bytes) -> void* {
        void* p = ws + off;
        off = (off + bytes + 255) & ~(size_t)255;
        return p;
    };
    float* dinv   = (float*)carve((size_t)N_NODES * 4);
    float* enorm  = (float*)carve((size_t)N_EDGES * 4);
    bf16*  nf     = (bf16*) carve((size_t)N_NODES * NF_DIM * 2);
    bf16*  h0     = (bf16*) carve((size_t)N_NODES * EMB * 2);
    bf16*  h1     = (bf16*) carve((size_t)N_NODES * H1_DIM * 2);
    float* tbuf   = (float*)carve((size_t)N_NODES * 128 * 4);
    float* agg    = (float*)carve((size_t)N_NODES * 128 * 4);
    bf16*  wsProj = (bf16*) carve((size_t)(NF_DIM/32) * (EMB/16)   * 512 * 2);
    bf16*  wsW1   = (bf16*) carve((size_t)(EMB/32)    * (H1_DIM/16)* 512 * 2);
    bf16*  wsW2   = (bf16*) carve((size_t)(H1_DIM/32) * (H2_DIM/16)* 512 * 2);
    float* pooled = (float*)carve((size_t)N_GRAPHS * 128 * 4);
    float* count  = (float*)carve((size_t)N_GRAPHS * 4);

    const int BLK = 256;
    auto cdiv = [](size_t a, size_t b) { return (int)((a + b - 1) / b); };

    // -------- weight swizzle (bf16 WMMA B layout) --------
    swizzle_w_kernel<<<cdiv((NF_DIM/32)*(EMB/16)*32, BLK),    BLK, 0, stream>>>(proj_W, wsProj, NF_DIM, EMB);
    swizzle_w_kernel<<<cdiv((EMB/32)*(H1_DIM/16)*32, BLK),    BLK, 0, stream>>>(W1, wsW1, EMB, H1_DIM);
    swizzle_w_kernel<<<cdiv((H1_DIM/32)*(H2_DIM/16)*32, BLK), BLK, 0, stream>>>(W2, wsW2, H1_DIM, H2_DIM);

    // -------- embeddings + degree norm --------
    gather_nf_kernel<<<cdiv((size_t)N_NODES * (NF_DIM/4), BLK), BLK, 0, stream>>>(
        x, depth, child, node_emb, depth_emb, child_emb, nf);
    deg_init_kernel<<<cdiv(N_NODES, BLK), BLK, 0, stream>>>(dinv);
    deg_edges_kernel<<<cdiv(N_EDGES, BLK), BLK, 0, stream>>>(e_dst, dinv);
    deg_finish_kernel<<<cdiv(N_NODES, BLK), BLK, 0, stream>>>(dinv);
    edge_norm_kernel<<<cdiv(N_EDGES, BLK), BLK, 0, stream>>>(e_src, e_dst, dinv, enorm);

    const int mBlocks = cdiv(N_NODES / 16, 8);   // 8 waves per block, 16 rows per wave
    const size_t nf128  = (size_t)N_NODES * 32;  // node feature rows as float4 groups
    const size_t ef128  = (size_t)N_EDGES * 32;

    // -------- GEMM1: h0 = relu(nf @ proj_W + proj_b), bf16 out --------
    gemm_bf16_kernel<<<dim3(mBlocks, EMB / 64), BLK, 0, stream>>>(
        nf, wsProj, nullptr, h0, proj_b, N_NODES, NF_DIM, EMB, 1);

    // -------- layer 1: t = h0 @ W1 ; aggregate ; finalize --------
    gemm_bf16_kernel<<<dim3(mBlocks, H1_DIM / 64), BLK, 0, stream>>>(
        h0, wsW1, tbuf, nullptr, nullptr, N_NODES, EMB, H1_DIM, 0);
    zero_f32v4_kernel<<<cdiv(nf128, BLK), BLK, 0, stream>>>((float4*)agg, nf128);
    edge_agg_kernel<<<cdiv(ef128, BLK), BLK, 0, stream>>>(e_src, e_dst, enorm, (const float4*)tbuf, agg);
    finalize_relu_bf16_kernel<<<cdiv(nf128, BLK), BLK, 0, stream>>>(
        (const float4*)agg, (const float4*)tbuf, dinv, (const float4*)b1, h1);

    // -------- layer 2: t = h1 @ W2 ; aggregate ; finalize + pool --------
    gemm_bf16_kernel<<<dim3(mBlocks, H2_DIM / 64), BLK, 0, stream>>>(
        h1, wsW2, tbuf, nullptr, nullptr, N_NODES, H1_DIM, H2_DIM, 0);
    zero_f32v4_kernel<<<cdiv(nf128, BLK), BLK, 0, stream>>>((float4*)agg, nf128);
    zero_f32v4_kernel<<<cdiv((size_t)N_GRAPHS * 32, BLK), BLK, 0, stream>>>((float4*)pooled, (size_t)N_GRAPHS * 32);
    zero_f32_kernel<<<1, 64, 0, stream>>>(count, N_GRAPHS);
    edge_agg_kernel<<<cdiv(ef128, BLK), BLK, 0, stream>>>(e_src, e_dst, enorm, (const float4*)tbuf, agg);
    count_nodes_kernel<<<cdiv(N_NODES, BLK), BLK, 0, stream>>>(batch, count);
    finalize_pool_kernel<<<cdiv(nf128, BLK), BLK, 0, stream>>>(
        (const float4*)agg, (const float4*)tbuf, dinv, (const float4*)b2, batch, pooled);

    // -------- classifier --------
    classify_kernel<<<1, 64, 0, stream>>>(pooled, count, clf_W, clf_b, out);
}
